// SAGENet_12154757448124
// MI455X (gfx1250) — matrix operand order repaired and checked by reading, not compiled, not run on hardware
//
#include <hip/hip_runtime.h>

// ---------------------------------------------------------------------------
// GraphSAGE 2-layer forward for MI455X (gfx1250, wave32, WMMA bf16 path)
//   layer: h = relu( mean_agg(x)@Wl1 + x@Wr1 + b1 )
//          o =       mean_agg(h)@Wl2 + h@Wr2 + b2
// GEMMs run on v_wmma_f32_16x16x32_bf16 (bf16 in, f32 accumulate).
// Each wave computes TWO 16x16 output tiles sharing one B fragment:
// halves B traffic and interleaves two independent WMMA chains.
// ---------------------------------------------------------------------------

#define N_NODES 100000
#define N_EDGES 600000
#define F_IN    128
#define HID     256

typedef __bf16 bf16_t;
typedef __attribute__((ext_vector_type(16))) __bf16 v16bf;
typedef __attribute__((ext_vector_type(8)))  __bf16 v8bf;
typedef __attribute__((ext_vector_type(8)))  float  v8f;

// ---------------- elementwise / prep kernels -------------------------------

__global__ void degree_kernel(const int* __restrict__ dst, float* __restrict__ cnt) {
    int e = blockIdx.x * blockDim.x + threadIdx.x;
    if (e < N_EDGES) atomicAdd(&cnt[dst[e]], 1.0f);
}

// W: f32 [K, HID] row-major  ->  Wt: bf16 [HID, K] (column-major access for WMMA B)
__global__ void transpose_w_kernel(const float* __restrict__ W, bf16_t* __restrict__ Wt, int K) {
    int idx = blockIdx.x * blockDim.x + threadIdx.x;
    if (idx >= K * HID) return;
    int k = idx / HID;
    int n = idx - k * HID;
    Wt[(size_t)n * K + k] = (bf16_t)W[idx];
}

__global__ void f32_to_bf16_kernel(const float* __restrict__ in, bf16_t* __restrict__ out,
                                   long long total) {
    long long i = (long long)blockIdx.x * blockDim.x + threadIdx.x;
    if (i < total) out[i] = (bf16_t)in[i];
}

// one block per edge; blockDim == feature width
__global__ void scatter_add_f32_kernel(const float* __restrict__ x,
                                       const int* __restrict__ src,
                                       const int* __restrict__ dst,
                                       float* __restrict__ agg) {
    int e = blockIdx.x;
    int f = threadIdx.x;
    int s = src[e], d = dst[e];
    atomicAdd(&agg[(size_t)d * blockDim.x + f], x[(size_t)s * blockDim.x + f]);
}

__global__ void scatter_add_bf16_kernel(const bf16_t* __restrict__ h,
                                        const int* __restrict__ src,
                                        const int* __restrict__ dst,
                                        float* __restrict__ agg) {
    int e = blockIdx.x;
    int f = threadIdx.x;
    int s = src[e], d = dst[e];
    atomicAdd(&agg[(size_t)d * blockDim.x + f], (float)h[(size_t)s * blockDim.x + f]);
}

// out[i] = bf16( agg[i] / max(cnt[row],1) ), row = i >> logF
__global__ void normalize_kernel(const float* __restrict__ agg, const float* __restrict__ cnt,
                                 bf16_t* __restrict__ out, int logF, long long total) {
    long long i = (long long)blockIdx.x * blockDim.x + threadIdx.x;
    if (i >= total) return;
    float c = cnt[i >> logF];
    float inv = (c > 1.0f) ? (1.0f / c) : 1.0f;
    out[i] = (bf16_t)(agg[i] * inv);
}

// ---------------- WMMA GEMM ------------------------------------------------
// A: bf16 [M, K] row-major.  Bt: bf16 [HID, K] (i.e. B column-major).
//
// A-fragment (16-bit A 16x32, ISA 7.12.2): lane<16 holds row M=lane,
//   VGPR0..3 = K 0..7, VGPR4..7 = K 16..23; lane>=16 holds row M=lane-16,
//   VGPR0..3 = K 8..15, VGPR4..7 = K 24..31.
// B-fragment (32x16): lane<16 holds col N=lane, K 0..7 then 16..23; lane>=16 holds
//   col N=lane-16, K 8..15 then 24..31 -> two 16B loads from Bt.

__device__ __forceinline__ v8f wmma_bf16(v16bf a, v16bf b, v8f c) {
    return __builtin_amdgcn_wmma_f32_16x16x32_bf16(
        /*neg_a=*/false, a, /*neg_b=*/false, b,
        /*c_mod=*/(short)0, c, /*reuse_a=*/false, /*reuse_b=*/false);
}

// Accumulate two M-tiles (rows m0.. and m0+16..) against one shared B panel.
template <int K>
__device__ __forceinline__ void mm_panel2(const bf16_t* __restrict__ A,
                                          const bf16_t* __restrict__ Bt,
                                          int m0, int n0, int lane,
                                          v8f& acc0, v8f& acc1) {
    const int half = lane >> 4;      // 0 or 1
    const int lr   = lane & 15;
    const bf16_t* __restrict__ arow0 = A + (size_t)(m0 + lr) * K;
    const bf16_t* __restrict__ arow1 = arow0 + (size_t)16 * K;
    const bf16_t* __restrict__ bcol  = Bt + (size_t)(n0 + lr) * K;
#pragma unroll
    for (int kb = 0; kb < K; kb += 32) {
        v16bf a0, a1, b;
        ((v8bf*)&a0)[0] = *(const v8bf*)(arow0 + kb +      half * 8);
        ((v8bf*)&a0)[1] = *(const v8bf*)(arow0 + kb + 16 + half * 8);
        ((v8bf*)&a1)[0] = *(const v8bf*)(arow1 + kb +      half * 8);
        ((v8bf*)&a1)[1] = *(const v8bf*)(arow1 + kb + 16 + half * 8);
        ((v8bf*)&b)[0] = *(const v8bf*)(bcol + kb +      half * 8);
        ((v8bf*)&b)[1] = *(const v8bf*)(bcol + kb + 16 + half * 8);
        acc0 = wmma_bf16(a0, b, acc0);   // two independent accumulator chains
        acc1 = wmma_bf16(a1, b, acc1);
    }
}

// D = Aa@Ba + Ab@Bb + bias ; RELU ? bf16 out : f32 out.
// grid.x = M/32 (2 row tiles per wave), block = 512 = 16 waves (col tiles).
template <int KA, int KB, int RELU>
__global__ void __launch_bounds__(512)
gemm_dual_kernel(const bf16_t* __restrict__ Aa, const bf16_t* __restrict__ Ba,
                 const bf16_t* __restrict__ Ab, const bf16_t* __restrict__ Bb,
                 const float* __restrict__ bias,
                 float* __restrict__ outF, bf16_t* __restrict__ outB) {
    const int lane = threadIdx.x & 31;
    const int wave = threadIdx.x >> 5;        // 0..15 -> column tile
    const int m0 = blockIdx.x * 32;
    const int n0 = wave * 16;

    v8f acc0 = {}, acc1 = {};
    mm_panel2<KA>(Aa, Ba, m0, n0, lane, acc0, acc1);
    mm_panel2<KB>(Ab, Bb, m0, n0, lane, acc0, acc1);

    // C/D layout: VGPR j, lanes 0-15 -> (M=j, N=lane); lanes 16-31 -> (M=j+8).
    const int half = lane >> 4;
    const int n = n0 + (lane & 15);
    const float bv = bias[n];
#pragma unroll
    for (int j = 0; j < 8; ++j) {
        const int m = m0 + j + half * 8;
        float v0 = acc0[j] + bv;
        float v1 = acc1[j] + bv;
        if (RELU) {
            v0 = v0 > 0.0f ? v0 : 0.0f;
            v1 = v1 > 0.0f ? v1 : 0.0f;
            outB[(size_t)m * HID + n]        = (bf16_t)v0;
            outB[(size_t)(m + 16) * HID + n] = (bf16_t)v1;
        } else {
            outF[(size_t)m * HID + n]        = v0;
            outF[(size_t)(m + 16) * HID + n] = v1;
        }
    }
}

// ---------------- host launcher --------------------------------------------

extern "C" void kernel_launch(void* const* d_in, const int* in_sizes, int n_in,
                              void* d_out, int out_size, void* d_ws, size_t ws_size,
                              hipStream_t stream) {
    (void)in_sizes; (void)n_in; (void)out_size; (void)ws_size;

    const float* x   = (const float*)d_in[0];
    const int*   ei  = (const int*)d_in[1];          // [2, E] row-major
    const int*   src = ei;
    const int*   dst = ei + N_EDGES;
    const float* Wl1 = (const float*)d_in[2];
    const float* Wr1 = (const float*)d_in[3];
    const float* b1  = (const float*)d_in[4];
    const float* Wl2 = (const float*)d_in[5];
    const float* Wr2 = (const float*)d_in[6];
    const float* b2  = (const float*)d_in[7];
    float* out = (float*)d_out;

    // workspace carve-out (256B aligned slots), ~230 MB total
    char* ws = (char*)d_ws;
    size_t off = 0;
    auto carve = [&](size_t bytes) -> void* {
        void* p = ws + off;
        off += (bytes + 255) & ~(size_t)255;
        return p;
    };
    float*  cnt    = (float*)carve((size_t)N_NODES * 4);
    float*  aggF   = (float*)carve((size_t)N_NODES * HID * 4);   // reused both layers
    bf16_t* x_bf   = (bf16_t*)carve((size_t)N_NODES * F_IN * 2);
    bf16_t* agg_bf = (bf16_t*)carve((size_t)N_NODES * HID * 2);
    bf16_t* h_bf   = (bf16_t*)carve((size_t)N_NODES * HID * 2);
    bf16_t* Wl1t   = (bf16_t*)carve((size_t)F_IN * HID * 2);
    bf16_t* Wr1t   = (bf16_t*)carve((size_t)F_IN * HID * 2);
    bf16_t* Wl2t   = (bf16_t*)carve((size_t)HID * HID * 2);
    bf16_t* Wr2t   = (bf16_t*)carve((size_t)HID * HID * 2);

    const long long nF1 = (long long)N_NODES * F_IN;   // 12.8M
    const long long nF2 = (long long)N_NODES * HID;    // 25.6M

    // ---- prep: degree counts, weight transposes, x -> bf16 ----
    hipMemsetAsync(cnt, 0, (size_t)N_NODES * 4, stream);
    hipMemsetAsync(aggF, 0, (size_t)nF1 * 4, stream);
    degree_kernel<<<(N_EDGES + 255) / 256, 256, 0, stream>>>(dst, cnt);
    transpose_w_kernel<<<(F_IN * HID + 255) / 256, 256, 0, stream>>>(Wl1, Wl1t, F_IN);
    transpose_w_kernel<<<(F_IN * HID + 255) / 256, 256, 0, stream>>>(Wr1, Wr1t, F_IN);
    transpose_w_kernel<<<(HID * HID + 255) / 256, 256, 0, stream>>>(Wl2, Wl2t, HID);
    transpose_w_kernel<<<(HID * HID + 255) / 256, 256, 0, stream>>>(Wr2, Wr2t, HID);
    f32_to_bf16_kernel<<<(unsigned)((nF1 + 255) / 256), 256, 0, stream>>>(x, x_bf, nF1);

    // ---- layer 1: mean aggregate (f32 atomics), GEMM in bf16 WMMA ----
    scatter_add_f32_kernel<<<N_EDGES, F_IN, 0, stream>>>(x, src, dst, aggF);
    normalize_kernel<<<(unsigned)((nF1 + 255) / 256), 256, 0, stream>>>(aggF, cnt, agg_bf, 7, nF1);
    gemm_dual_kernel<F_IN, F_IN, 1><<<N_NODES / 32, 512, 0, stream>>>(
        agg_bf, Wl1t, x_bf, Wr1t, b1, nullptr, h_bf);

    // ---- layer 2 ----
    hipMemsetAsync(aggF, 0, (size_t)nF2 * 4, stream);
    scatter_add_bf16_kernel<<<N_EDGES, HID, 0, stream>>>(h_bf, src, dst, aggF);
    normalize_kernel<<<(unsigned)((nF2 + 255) / 256), 256, 0, stream>>>(aggF, cnt, agg_bf, 8, nF2);
    gemm_dual_kernel<HID, HID, 0><<<N_NODES / 32, 512, 0, stream>>>(
        agg_bf, Wl2t, h_bf, Wr2t, b2, out, nullptr);
}
